// FMVectorField_83829171683344
// MI455X (gfx1250) — matrix operand order repaired and checked by reading, not compiled, not run on hardware
//
#include <hip/hip_runtime.h>
#include <math.h>

// ---------------------------------------------------------------------------
// FMVectorField forward for MI455X (gfx1250, wave32, WMMA).
// Heavy per-edge message GVP uses v_wmma_f32_16x16x32_f16 (16 edges / wave).
// ---------------------------------------------------------------------------

typedef _Float16 v16h __attribute__((ext_vector_type(16)));
typedef float    v8f  __attribute__((ext_vector_type(8)));

struct GvpP { const float *Wg, *Wh, *Ws, *Wv, *bg, *bs; };

__device__ __forceinline__ float siluf(float x) { return x / (1.0f + expf(-x)); }
__device__ __forceinline__ float sigmf(float x) { return 1.0f / (1.0f + expf(-x)); }

__device__ __forceinline__ void layernorm64(float* x, const float* g, const float* b) {
    float mu = 0.0f;
    #pragma unroll
    for (int j = 0; j < 64; ++j) mu += x[j];
    mu *= (1.0f / 64.0f);
    float var = 0.0f;
    #pragma unroll
    for (int j = 0; j < 64; ++j) { float d = x[j] - mu; var += d * d; }
    var *= (1.0f / 64.0f);
    float rs = rsqrtf(var + 1e-5f);
    #pragma unroll
    for (int j = 0; j < 64; ++j) x[j] = (x[j] - mu) * rs * g[j] + b[j];
}

// ------------------------- small utility kernels ---------------------------

__global__ void zero_kernel(float* p, long n) {
    long i = (long)blockIdx.x * blockDim.x + threadIdx.x;
    if (i < n) p[i] = 0.0f;
}

__global__ void copy_kernel(float* dst, const float* src, long n) {
    long i = (long)blockIdx.x * blockDim.x + threadIdx.x;
    if (i < n) dst[i] = src[i];
}

// ------------------------------ embedding ----------------------------------
// s = LN(silu(silu([type_emb[h], t[bidx]] @ W1 + b1) @ W2 + b2)); v = 0

__global__ void embed_kernel(const int* __restrict__ ht, const int* __restrict__ bidx,
                             const float* __restrict__ t, const float* __restrict__ TE,
                             const float* __restrict__ W1, const float* __restrict__ b1,
                             const float* __restrict__ W2, const float* __restrict__ b2,
                             const float* __restrict__ lnb, const float* __restrict__ lng,
                             float* __restrict__ sF, float* __restrict__ vF, int N) {
    int n = blockIdx.x * blockDim.x + threadIdx.x;
    if (n >= N) return;
    float z[65];
    const float* emb = TE + (size_t)ht[n] * 64;
    #pragma unroll
    for (int i = 0; i < 64; ++i) z[i] = emb[i];
    z[64] = t[bidx[n]];
    float z1[64];
    for (int j = 0; j < 64; ++j) {
        float acc = b1[j];
        for (int k = 0; k < 65; ++k) acc += z[k] * W1[k * 64 + j];
        z1[j] = siluf(acc);
    }
    float z2[64];
    for (int j = 0; j < 64; ++j) {
        float acc = b2[j];
        for (int k = 0; k < 64; ++k) acc += z1[k] * W2[k * 64 + j];
        z2[j] = siluf(acc);
    }
    layernorm64(z2, lng, lnb);
    float* so = sF + (size_t)n * 64;
    #pragma unroll
    for (int j = 0; j < 64; ++j) so[j] = z2[j];
    float* vo = vF + (size_t)n * 48;
    #pragma unroll
    for (int j = 0; j < 48; ++j) vo[j] = 0.0f;
}

// -------------------------- edge message kernel (WMMA) ---------------------
// One wave = 16 edges. s_cat[16 x 161->192] @ Ws[161x64] via f16 WMMA,
// then gate: s_out[16x64] @ Wg[64x16] via f16 WMMA. f32 atomics into agg.

#define MSG_NW 4  // waves per block (128 threads)

__global__ __launch_bounds__(128) void msg_kernel(
    const float* __restrict__ sSrc, const float* __restrict__ vSrc, const float* __restrict__ xSrc,
    const float* __restrict__ sDst, const float* __restrict__ xDst,
    const int* __restrict__ src, const int* __restrict__ dst, int E,
    const float* __restrict__ Wh, const float* __restrict__ Ws, const float* __restrict__ bs,
    const float* __restrict__ Wv, const float* __restrict__ Wg, const float* __restrict__ bg,
    float* __restrict__ aggS, float* __restrict__ aggV, float* __restrict__ cnt) {
    __shared__ _Float16 WsT[64][192];   // [n][k], k >= 161 zero padded
    __shared__ _Float16 WgT[16][64];    // [u][k]
    __shared__ float    vhBuf[MSG_NW][16][17][3];
    __shared__ _Float16 sOutH[MSG_NW][16][64];
    __shared__ int      srcIdx[MSG_NW][16];
    __shared__ int      dstIdx[MSG_NW][16];
    __shared__ float    xdBuf[MSG_NW][16][3];
    __shared__ float    dBuf[MSG_NW][16];

    const int tid = threadIdx.x;
    for (int i = tid; i < 64 * 192; i += 128) {
        int n = i / 192, k = i - n * 192;
        WsT[n][k] = (k < 161) ? (_Float16)Ws[(size_t)k * 64 + n] : (_Float16)0.0f;
    }
    for (int i = tid; i < 16 * 64; i += 128) {
        int u = i >> 6, k = i & 63;
        WgT[u][k] = (_Float16)Wg[(size_t)k * 16 + u];
    }

    const int wave = tid >> 5, lane = tid & 31;
    const int laneM = lane & 15, hi = lane >> 4;
    const long tileBase = ((long)blockIdx.x * MSG_NW + wave) * 16;

    // phase 1: indices + geometry (rbf distance + unit diff)
    if (lane < 16) {
        long e = tileBase + laneM;
        int si = 0, di = 0;
        if (e < (long)E) { si = src[e]; di = dst[e]; }
        srcIdx[wave][laneM] = si;
        dstIdx[wave][laneM] = di;
        float dx = xSrc[si * 3 + 0] - xDst[di * 3 + 0];
        float dy = xSrc[si * 3 + 1] - xDst[di * 3 + 1];
        float dz = xSrc[si * 3 + 2] - xDst[di * 3 + 2];
        float d = sqrtf(dx * dx + dy * dy + dz * dz + 1e-8f) + 1e-8f;
        dBuf[wave][laneM] = d;
        float id = 1.0f / d;
        xdBuf[wave][laneM][0] = dx * id;
        xdBuf[wave][laneM][1] = dy * id;
        xdBuf[wave][laneM][2] = dz * id;
    }
    __syncthreads();

    // phase 2: vh = Wh^T v_in  (2 lanes per edge; lane handles h-range)
    {
        int si = srcIdx[wave][laneM];
        float vin[17][3];
        const float* vp = vSrc + (size_t)si * 48;
        #pragma unroll
        for (int v = 0; v < 16; ++v) {
            vin[v][0] = vp[v * 3 + 0]; vin[v][1] = vp[v * 3 + 1]; vin[v][2] = vp[v * 3 + 2];
        }
        vin[16][0] = xdBuf[wave][laneM][0];
        vin[16][1] = xdBuf[wave][laneM][1];
        vin[16][2] = xdBuf[wave][laneM][2];
        int h0 = hi ? 9 : 0, h1 = hi ? 17 : 9;
        for (int h = h0; h < h1; ++h) {
            float a0 = 0.f, a1 = 0.f, a2 = 0.f;
            #pragma unroll
            for (int v = 0; v < 17; ++v) {
                float w = Wh[v * 17 + h];
                a0 += vin[v][0] * w; a1 += vin[v][1] * w; a2 += vin[v][2] * w;
            }
            vhBuf[wave][laneM][h][0] = a0;
            vhBuf[wave][laneM][h][1] = a1;
            vhBuf[wave][laneM][h][2] = a2;
        }
    }
    __syncthreads();

    const int   mySi = srcIdx[wave][laneM];
    const int   myDi = dstIdx[wave][laneM];
    const float myD  = dBuf[wave][laneM];

    // s_cat element for my A-row (edge = laneM within tile)
    auto aval = [&](int k) -> float {
        if (k < 64)  return sSrc[(size_t)mySi * 64 + k];
        if (k < 128) return sDst[(size_t)myDi * 64 + (k - 64)];
        if (k < 144) {
            float mu = (float)(k - 128) * (20.0f / 15.0f);
            float tt = (myD - mu) * 0.8f;  // / sigma (1.25)
            return expf(-tt * tt);
        }
        if (k < 161) {
            int h = k - 144;
            float x = vhBuf[wave][laneM][h][0];
            float y = vhBuf[wave][laneM][h][1];
            float z = vhBuf[wave][laneM][h][2];
            return sqrtf(x * x + y * y + z * z + 1e-8f);
        }
        return 0.0f;
    };

    // WMMA #1: s_out_tile[16x64] = s_cat_tile[16x192] @ WsT  (K padded to 192)
    v8f acc0 = {}, acc1 = {}, acc2 = {}, acc3 = {};
    for (int kc = 0; kc < 6; ++kc) {
        v16h a;
        int kb0 = kc * 32 + (hi ? 16 : 0);
        int kb1 = kc * 32 + (hi ? 24 : 8);
        #pragma unroll
        for (int j = 0; j < 8; ++j) {
            a[j]     = (_Float16)aval(kb0 + j);
            a[j + 8] = (_Float16)aval(kb1 + j);
        }
        int kb = kc * 32 + hi * 16;
        v16h b;
        #pragma unroll
        for (int j = 0; j < 16; ++j) b[j] = WsT[laneM][kb + j];
        acc0 = __builtin_amdgcn_wmma_f32_16x16x32_f16(false, a, false, b, (short)0, acc0, false, false);
        #pragma unroll
        for (int j = 0; j < 16; ++j) b[j] = WsT[16 + laneM][kb + j];
        acc1 = __builtin_amdgcn_wmma_f32_16x16x32_f16(false, a, false, b, (short)0, acc1, false, false);
        #pragma unroll
        for (int j = 0; j < 16; ++j) b[j] = WsT[32 + laneM][kb + j];
        acc2 = __builtin_amdgcn_wmma_f32_16x16x32_f16(false, a, false, b, (short)0, acc2, false, false);
        #pragma unroll
        for (int j = 0; j < 16; ++j) b[j] = WsT[48 + laneM][kb + j];
        acc3 = __builtin_amdgcn_wmma_f32_16x16x32_f16(false, a, false, b, (short)0, acc3, false, false);
    }

    // bias + silu; atomic aggS; stage f16 s_out for gate WMMA
    v8f accs[4] = {acc0, acc1, acc2, acc3};
    #pragma unroll
    for (int nt = 0; nt < 4; ++nt) {
        int n = nt * 16 + laneM;
        float bias = bs[n];
        #pragma unroll
        for (int v = 0; v < 8; ++v) {
            int M = v + 8 * hi;
            float x = accs[nt][v] + bias;
            float so = siluf(x);
            sOutH[wave][M][n] = (_Float16)so;
            long e = tileBase + M;
            if (e < (long)E) atomicAdd(&aggS[(size_t)dstIdx[wave][M] * 64 + n], so);
        }
    }
    if (lane < 16 && (tileBase + laneM) < (long)E)
        atomicAdd(&cnt[dstIdx[wave][laneM]], 1.0f);
    __syncthreads();

    // WMMA #2: gate_logits[16x16] = s_out_tile[16x64] @ WgT
    v8f gacc = {};
    #pragma unroll
    for (int kc = 0; kc < 2; ++kc) {
        v16h a;
        int kb0 = kc * 32 + (hi ? 16 : 0);
        int kb1 = kc * 32 + (hi ? 24 : 8);
        #pragma unroll
        for (int j = 0; j < 8; ++j) {
            a[j]     = sOutH[wave][laneM][kb0 + j];
            a[j + 8] = sOutH[wave][laneM][kb1 + j];
        }
        v16h b;
        int kb = kc * 32 + hi * 16;
        #pragma unroll
        for (int j = 0; j < 16; ++j) b[j] = WgT[laneM][kb + j];
        gacc = __builtin_amdgcn_wmma_f32_16x16x32_f16(false, a, false, b, (short)0, gacc, false, false);
    }

    // vm = (Wv^T vh) * sigmoid(gate); atomic aggV. lane owns channel u=laneM.
    float bgu = bg[laneM];
    #pragma unroll
    for (int v = 0; v < 8; ++v) {
        int M = v + 8 * hi;
        long e = tileBase + M;
        if (e >= (long)E) continue;
        float gate = sigmf(gacc[v] + bgu);
        float u0 = 0.f, u1 = 0.f, u2 = 0.f;
        #pragma unroll
        for (int h = 0; h < 17; ++h) {
            float w = Wv[h * 16 + laneM];
            u0 += vhBuf[wave][M][h][0] * w;
            u1 += vhBuf[wave][M][h][1] * w;
            u2 += vhBuf[wave][M][h][2] * w;
        }
        int di = dstIdx[wave][M];
        size_t base = (size_t)di * 48 + laneM * 3;
        atomicAdd(&aggV[base + 0], u0 * gate);
        atomicAdd(&aggV[base + 1], u1 * gate);
        atomicAdd(&aggV[base + 2], u2 * gate);
    }
}

// ------------------------- node-update GVP (scalar) ------------------------

__device__ void gvp16(float s[64], float v[16][3], const GvpP p, bool sigGate) {
    float vh[16][3];
    for (int h = 0; h < 16; ++h) {
        float a0 = 0.f, a1 = 0.f, a2 = 0.f;
        #pragma unroll
        for (int q = 0; q < 16; ++q) {
            float w = p.Wh[q * 16 + h];
            a0 += v[q][0] * w; a1 += v[q][1] * w; a2 += v[q][2] * w;
        }
        vh[h][0] = a0; vh[h][1] = a1; vh[h][2] = a2;
    }
    float scat[80];
    #pragma unroll
    for (int j = 0; j < 64; ++j) scat[j] = s[j];
    #pragma unroll
    for (int h = 0; h < 16; ++h)
        scat[64 + h] = sqrtf(vh[h][0] * vh[h][0] + vh[h][1] * vh[h][1] + vh[h][2] * vh[h][2] + 1e-8f);
    float sout[64];
    for (int j = 0; j < 64; ++j) {
        float acc = p.bs[j];
        for (int k = 0; k < 80; ++k) acc += scat[k] * p.Ws[k * 64 + j];
        sout[j] = siluf(acc);
    }
    for (int u = 0; u < 16; ++u) {
        float gl = p.bg[u];
        for (int j = 0; j < 64; ++j) gl += sout[j] * p.Wg[j * 16 + u];
        float gate = sigGate ? sigmf(gl) : gl;
        float b0 = 0.f, b1 = 0.f, b2 = 0.f;
        #pragma unroll
        for (int h = 0; h < 16; ++h) {
            float w = p.Wv[h * 16 + u];
            b0 += vh[h][0] * w; b1 += vh[h][1] * w; b2 += vh[h][2] * w;
        }
        v[u][0] = b0 * gate; v[u][1] = b1 * gate; v[u][2] = b2 * gate;
    }
    #pragma unroll
    for (int j = 0; j < 64; ++j) s[j] = sout[j];
}

__global__ void update_kernel(float* __restrict__ sF, float* __restrict__ vF,
                              const float* __restrict__ aggS, const float* __restrict__ aggV,
                              const float* __restrict__ cnt,
                              const float* __restrict__ ln1g, const float* __restrict__ ln1b,
                              const float* __restrict__ ln2g, const float* __restrict__ ln2b,
                              GvpP p, int N) {
    int n = blockIdx.x * blockDim.x + threadIdx.x;
    if (n >= N) return;
    float c = cnt[n]; if (c < 1.0f) c = 1.0f;
    float inv = 1.0f / c;
    float s1[64];
    #pragma unroll
    for (int j = 0; j < 64; ++j) s1[j] = sF[(size_t)n * 64 + j] + aggS[(size_t)n * 64 + j] * inv;
    layernorm64(s1, ln1g, ln1b);
    float v1[16][3];
    #pragma unroll
    for (int q = 0; q < 16; ++q)
        for (int i = 0; i < 3; ++i)
            v1[q][i] = vF[(size_t)n * 48 + q * 3 + i] + aggV[(size_t)n * 48 + q * 3 + i] * inv;
    float st[64], vt[16][3];
    #pragma unroll
    for (int j = 0; j < 64; ++j) st[j] = s1[j];
    #pragma unroll
    for (int q = 0; q < 16; ++q) { vt[q][0] = v1[q][0]; vt[q][1] = v1[q][1]; vt[q][2] = v1[q][2]; }
    gvp16(st, vt, p, true);
    #pragma unroll
    for (int j = 0; j < 64; ++j) s1[j] += st[j];
    layernorm64(s1, ln2g, ln2b);
    #pragma unroll
    for (int j = 0; j < 64; ++j) sF[(size_t)n * 64 + j] = s1[j];
    #pragma unroll
    for (int q = 0; q < 16; ++q)
        for (int i = 0; i < 3; ++i)
            vF[(size_t)n * 48 + q * 3 + i] = v1[q][i] + vt[q][i];
}

// --------------------- position update (3 chained GVPs) --------------------

__global__ void pos_kernel(const float* __restrict__ sF, const float* __restrict__ vF,
                           float* __restrict__ x, GvpP g0, GvpP g1, GvpP g2, int N) {
    int n = blockIdx.x * blockDim.x + threadIdx.x;
    if (n >= N) return;
    float s[64], v[16][3];
    #pragma unroll
    for (int j = 0; j < 64; ++j) s[j] = sF[(size_t)n * 64 + j];
    #pragma unroll
    for (int q = 0; q < 16; ++q)
        for (int i = 0; i < 3; ++i) v[q][i] = vF[(size_t)n * 48 + q * 3 + i];
    gvp16(s, v, g0, true);
    gvp16(s, v, g1, true);
    // last GVP: v_out = 1, identity gate
    float vh[16][3];
    for (int h = 0; h < 16; ++h) {
        float a0 = 0.f, a1 = 0.f, a2 = 0.f;
        #pragma unroll
        for (int q = 0; q < 16; ++q) {
            float w = g2.Wh[q * 16 + h];
            a0 += v[q][0] * w; a1 += v[q][1] * w; a2 += v[q][2] * w;
        }
        vh[h][0] = a0; vh[h][1] = a1; vh[h][2] = a2;
    }
    float scat[80];
    #pragma unroll
    for (int j = 0; j < 64; ++j) scat[j] = s[j];
    #pragma unroll
    for (int h = 0; h < 16; ++h)
        scat[64 + h] = sqrtf(vh[h][0] * vh[h][0] + vh[h][1] * vh[h][1] + vh[h][2] * vh[h][2] + 1e-8f);
    float gl = g2.bg[0];
    for (int j = 0; j < 64; ++j) {
        float acc = g2.bs[j];
        for (int k = 0; k < 80; ++k) acc += scat[k] * g2.Ws[k * 64 + j];
        gl += siluf(acc) * g2.Wg[j];
    }
    float d0 = 0.f, d1 = 0.f, d2 = 0.f;
    #pragma unroll
    for (int h = 0; h < 16; ++h) {
        float w = g2.Wv[h];
        d0 += vh[h][0] * w; d1 += vh[h][1] * w; d2 += vh[h][2] * w;
    }
    x[n * 3 + 0] += d0 * gl;
    x[n * 3 + 1] += d1 * gl;
    x[n * 3 + 2] += d2 * gl;
}

// ------------------------------- readout -----------------------------------

__global__ void readout_kernel(const float* __restrict__ sF,
                               const float* __restrict__ W1, const float* __restrict__ b1,
                               const float* __restrict__ W2, const float* __restrict__ b2,
                               float* __restrict__ out, int N) {
    int n = blockIdx.x * blockDim.x + threadIdx.x;
    if (n >= N) return;
    float z1[64];
    for (int j = 0; j < 64; ++j) {
        float acc = b1[j];
        for (int k = 0; k < 64; ++k) acc += sF[(size_t)n * 64 + k] * W1[k * 64 + j];
        z1[j] = siluf(acc);
    }
    #pragma unroll
    for (int o = 0; o < 8; ++o) {
        float acc = b2[o];
        for (int j = 0; j < 64; ++j) acc += z1[j] * W2[j * 8 + o];
        out[(size_t)n * 8 + o] = acc;
    }
}

// ------------------------------- launcher ----------------------------------
// Param leaf order assumes JAX pytree flatten: dict keys sorted, lists in order.

extern "C" void kernel_launch(void* const* d_in, const int* in_sizes, int n_in,
                              void* d_out, int out_size, void* d_ws, size_t ws_size,
                              hipStream_t stream) {
    if (n_in < 219) return;
    const int nph = in_sizes[0];
    const int npr = in_sizes[1];
    const int Eff = in_sizes[7], Epf = in_sizes[9], Efp = in_sizes[11], Epp = in_sizes[13];

    #define FPX(i) ((const float*)d_in[(i)])
    #define IPX(i) ((const int*)d_in[(i)])

    float* ws = (float*)d_ws;
    size_t off = 0;
    float* sPh = ws + off; off += (size_t)nph * 64;
    float* sPr = ws + off; off += (size_t)npr * 64;
    float* vPh = ws + off; off += (size_t)nph * 48;
    float* vPr = ws + off; off += (size_t)npr * 48;
    float* aggBase = ws + off;
    float* aSPh = ws + off; off += (size_t)nph * 64;
    float* aSPr = ws + off; off += (size_t)npr * 64;
    float* aVPh = ws + off; off += (size_t)nph * 48;
    float* aVPr = ws + off; off += (size_t)npr * 48;
    float* cPh  = ws + off; off += nph;
    float* cPr  = ws + off; off += npr;
    float* xPh  = ws + off; off += (size_t)nph * 3;
    const size_t aggLen = (size_t)(nph + npr) * 112 + nph + npr;

    // working copy of pharmacophore positions (updated by pos_kernel)
    copy_kernel<<<(nph * 3 + 255) / 256, 256, 0, stream>>>(xPh, FPX(2), (long)nph * 3);

    // node embedding (pharm base 147, prot base 153; order W1,W2,b1,b2,ln.b,ln.g)
    embed_kernel<<<(nph + 127) / 128, 128, 0, stream>>>(IPX(0), IPX(5), FPX(4), FPX(163),
        FPX(147), FPX(149), FPX(148), FPX(150), FPX(151), FPX(152), sPh, vPh, nph);
    embed_kernel<<<(npr + 127) / 128, 128, 0, stream>>>(IPX(1), IPX(6), FPX(4), FPX(164),
        FPX(153), FPX(155), FPX(154), FPX(156), FPX(157), FPX(158), sPr, vPr, npr);

    for (int li = 0; li < 3; ++li) {
        const int CB = 15 + 44 * li;  // conv base: ln1(4), ln2(4), msg ff/fp/pf/pp(24), upd(12)
        zero_kernel<<<(int)((aggLen + 255) / 256), 256, 0, stream>>>(aggBase, (long)aggLen);

        auto msg = [&](int mb, const float* sS, const float* vS, const float* xS,
                       const float* sD, const float* xD, const int* si, const int* di, int E,
                       float* aS, float* aV, float* cc) {
            int blocks = (E + MSG_NW * 16 - 1) / (MSG_NW * 16);
            // GVP leaf order per sorted dict: Wg(+0), Wh(+1), Ws(+2), Wv(+3), bg(+4), bs(+5)
            msg_kernel<<<blocks, 128, 0, stream>>>(sS, vS, xS, sD, xD, si, di, E,
                FPX(mb + 1), FPX(mb + 2), FPX(mb + 5), FPX(mb + 3), FPX(mb + 0), FPX(mb + 4),
                aS, aV, cc);
        };
        // reference ETYPES order ff, pf, fp, pp; msg dict sorted ff(+8), fp(+14), pf(+20), pp(+26)
        msg(CB + 8,  sPh, vPh, xPh,    sPh, xPh,    IPX(7),  IPX(8),  Eff, aSPh, aVPh, cPh);
        msg(CB + 20, sPr, vPr, FPX(3), sPh, xPh,    IPX(9),  IPX(10), Epf, aSPh, aVPh, cPh);
        msg(CB + 14, sPh, vPh, xPh,    sPr, FPX(3), IPX(11), IPX(12), Efp, aSPr, aVPr, cPr);
        msg(CB + 26, sPr, vPr, FPX(3), sPr, FPX(3), IPX(13), IPX(14), Epp, aSPr, aVPr, cPr);

        GvpP uph = { FPX(CB + 32), FPX(CB + 33), FPX(CB + 34), FPX(CB + 35), FPX(CB + 36), FPX(CB + 37) };
        GvpP upr = { FPX(CB + 38), FPX(CB + 39), FPX(CB + 40), FPX(CB + 41), FPX(CB + 42), FPX(CB + 43) };
        update_kernel<<<(nph + 127) / 128, 128, 0, stream>>>(sPh, vPh, aSPh, aVPh, cPh,
            FPX(CB + 1), FPX(CB + 0), FPX(CB + 5), FPX(CB + 4), uph, nph);
        update_kernel<<<(npr + 127) / 128, 128, 0, stream>>>(sPr, vPr, aSPr, aVPr, cPr,
            FPX(CB + 3), FPX(CB + 2), FPX(CB + 7), FPX(CB + 6), upr, npr);

        const int UB = 165 + 18 * li;
        GvpP g0 = { FPX(UB + 0),  FPX(UB + 1),  FPX(UB + 2),  FPX(UB + 3),  FPX(UB + 4),  FPX(UB + 5)  };
        GvpP g1 = { FPX(UB + 6),  FPX(UB + 7),  FPX(UB + 8),  FPX(UB + 9),  FPX(UB + 10), FPX(UB + 11) };
        GvpP g2 = { FPX(UB + 12), FPX(UB + 13), FPX(UB + 14), FPX(UB + 15), FPX(UB + 16), FPX(UB + 17) };
        pos_kernel<<<(nph + 127) / 128, 128, 0, stream>>>(sPh, vPh, xPh, g0, g1, g2, nph);
    }

    // outputs: logits [nph x 8] then positions [nph x 3]
    readout_kernel<<<(nph + 127) / 128, 128, 0, stream>>>(sPh, FPX(159), FPX(161), FPX(160), FPX(162),
                                                          (float*)d_out, nph);
    copy_kernel<<<(nph * 3 + 255) / 256, 256, 0, stream>>>((float*)d_out + (size_t)nph * 8, xPh,
                                                           (long)nph * 3);
    #undef FPX
    #undef IPX
}
